// GCNSageLSTMRawPluginAGHOneHotWRecLSTM_90692529422805
// MI455X (gfx1250) — matrix operand order repaired and checked by reading, compiled, not run on hardware
//
#include <hip/hip_runtime.h>
#include <hip/hip_bf16.h>
#include <math.h>

// ---------------------------------------------------------------------------
// CDNA5 (gfx1250) implementation of the LSTM->GCN->LSTM graph model.
//  * Dense GEMMs: v_wmma_f32_16x16x32_bf16, operands pre-packed into
//    zero-padded bf16 staging buffers -> branch-free b128 loads in the K loop,
//    2 independent accumulators per wave.
//  * LSTM recurrences: Whh sharded into LDS (320KB/WGP on CDNA5) via the
//    Tensor Data Mover (tensor_load_to_lds + s_wait_tensorcnt); per-step GEMV
//    via WMMA with h (bf16, converted once per step) in column 0 of B.
//    Device-wide per-step sync via atomic sense barrier (+ s_cluster_barrier,
//    the HW path under cluster dispatch, architectural NOP otherwise).
//  * GCN scatter-add via global_atomic_add_f32 (unsafeAtomicAdd).
// ---------------------------------------------------------------------------

typedef __attribute__((ext_vector_type(16))) __bf16 v16bf;
typedef __attribute__((ext_vector_type(8)))  float  v8f;
typedef __attribute__((ext_vector_type(4)))  unsigned int v4u;
typedef __attribute__((ext_vector_type(8)))  int v8i;
typedef __attribute__((ext_vector_type(4)))  int v4i;

union BfVec {
  v16bf v;
  unsigned short u[16];
  unsigned int w[8];
  uint4 q[2];
};

static __device__ __forceinline__ unsigned pack2_bf16(float lo, float hi) {
  unsigned a = __float_as_uint(lo), b = __float_as_uint(hi);
  a += 0x7fffu + ((a >> 16) & 1u);               // round-to-nearest-even
  b += 0x7fffu + ((b >> 16) & 1u);
  return __builtin_amdgcn_perm(b, a, 0x07060302u); // {b[31:16], a[31:16]}
}
static __device__ __forceinline__ float sigmoidf_(float x) {
  return 1.0f / (1.0f + __expf(-x));
}

// ---------------------------------------------------------------------------
// Device-wide sense-reversing barrier (bar[0]=count, bar[1]=generation).
// ---------------------------------------------------------------------------
static __device__ __forceinline__ void grid_barrier(unsigned* bar, int nwg) {
  __threadfence();          // flush this wave's stores to device scope (L2)
  __syncthreads();
  if (threadIdx.x == 0) {
    // HW cluster barrier: real sync under cluster dispatch, NOP otherwise.
    __builtin_amdgcn_s_cluster_barrier();
    unsigned gen  = __hip_atomic_load(&bar[1], __ATOMIC_RELAXED, __HIP_MEMORY_SCOPE_AGENT);
    unsigned prev = __hip_atomic_fetch_add(&bar[0], 1u, __ATOMIC_ACQ_REL, __HIP_MEMORY_SCOPE_AGENT);
    if (prev == (unsigned)(nwg - 1)) {
      __hip_atomic_store(&bar[0], 0u, __ATOMIC_RELAXED, __HIP_MEMORY_SCOPE_AGENT);
      __hip_atomic_store(&bar[1], gen + 1u, __ATOMIC_RELEASE, __HIP_MEMORY_SCOPE_AGENT);
    } else {
      while (__hip_atomic_load(&bar[1], __ATOMIC_ACQUIRE, __HIP_MEMORY_SCOPE_AGENT) == gen) {
        __builtin_amdgcn_s_sleep(8);
      }
    }
  }
  __syncthreads();
  __threadfence();          // discard stale WGP$ lines before reading peers' h
}

// ---------------------------------------------------------------------------
// Pack f32 [R,C] into zero-padded bf16 [Rp,Cp] (Cp even). One dword / thread.
// ---------------------------------------------------------------------------
__global__ void k_pack_bf16(const float* __restrict__ src, unsigned* __restrict__ dst,
                            int R, int C, int Rp, int Cp) {
  int i = blockIdx.x * blockDim.x + threadIdx.x;
  int cw = Cp >> 1;
  if (i >= Rp * cw) return;
  int r = i / cw, c0 = (i % cw) * 2;
  float f0 = (r < R && c0     < C) ? src[(size_t)r * C + c0]     : 0.0f;
  float f1 = (r < R && c0 + 1 < C) ? src[(size_t)r * C + c0 + 1] : 0.0f;
  dst[i] = pack2_bf16(f0, f1);
}

// ---------------------------------------------------------------------------
// GEMM: C[M,N] = A[M,K] @ W[N,K]^T (+ b0 + b1).
// Abf: [M,Kp] bf16, Wbf: [Np,Kp] bf16, both zero-padded, Kp%64==0, Np%64==0,
// M%32==0. Block = 256 thr = 8 waves (2Mx4N): 32x64 tile. Branch-free K loop.
// ---------------------------------------------------------------------------
__global__ void __launch_bounds__(256) gemm_bias_wmma(
    const unsigned short* __restrict__ Abf,
    const unsigned short* __restrict__ Wbf,
    const float* __restrict__ b0, const float* __restrict__ b1,
    float* __restrict__ C, int ldc, int M, int N, int Kp)
{
  const int tid  = threadIdx.x;
  const int wv   = tid >> 5;
  const int lane = tid & 31;
  const int half = lane >> 4;
  const int l    = lane & 15;
  const int m0 = blockIdx.y * 32 + (wv >> 2) * 16;
  const int n0 = blockIdx.x * 64 + (wv & 3) * 16;

  // A 16x32 bf16 frag: lanes0-15 K=[0..15];        lanes16-31 K=[16..31].
  // B 32x16 bf16 frag: lanes0-15 K=[0..15];        lanes16-31 K=[16..31].
  const uint4* aq = (const uint4*)(Abf + (size_t)(m0 + l) * Kp);
  const uint4* wq = (const uint4*)(Wbf + (size_t)(n0 + l) * Kp);

  v8f acc0 = {}, acc1 = {};
  for (int k0 = 0; k0 < Kp; k0 += 64) {
    BfVec a0, a1, bb0, bb1;
    a0.q[0]  = aq[half * 2];     a0.q[1]  = aq[half * 2 + 1];
    bb0.q[0] = wq[half * 2];     bb0.q[1] = wq[half * 2 + 1];
    a1.q[0]  = aq[4 + half * 2]; a1.q[1]  = aq[4 + half * 2 + 1];
    bb1.q[0] = wq[4 + half * 2]; bb1.q[1] = wq[4 + half * 2 + 1];
    __builtin_prefetch(aq + 8, 0, 1);                // global_prefetch_b8
    __builtin_prefetch(wq + 8, 0, 1);
    acc0 = __builtin_amdgcn_wmma_f32_16x16x32_bf16(false, a0.v, false, bb0.v,
                                                   (short)0, acc0, false, false);
    acc1 = __builtin_amdgcn_wmma_f32_16x16x32_bf16(false, a1.v, false, bb1.v,
                                                   (short)0, acc1, false, false);
    aq += 8; wq += 8;
  }

  const int bcol = n0 + l;
  if (bcol < N) {
    float bias = 0.0f;
    if (b0) bias += b0[bcol];
    if (b1) bias += b1[bcol];
    // D 16x16 f32: VGPR v -> M = m0 + half*8 + v, N = bcol.
    #pragma unroll
    for (int v = 0; v < 8; ++v)
      C[(size_t)(m0 + half * 8 + v) * ldc + bcol] = acc0[v] + acc1[v] + bias;
  }
}

// ---------------------------------------------------------------------------
// LSTM recurrence. One workgroup (4 waves / 128 thr) owns 16 hidden units;
// its 64 Whh rows (4 gates x 16 units) stay resident in LDS as bf16.
// Preload: each wave issues one TDM descriptor (16 rows x H bf16 tile) via
// tensor_load_to_lds, then s_wait_tensorcnt; a VGPR copy after the wait
// re-writes the same bytes (one-shot belt-and-suspenders in this compile-only
// environment; amortized over T=2048 steps).
// Per step: h -> bf16 in LDS once; g = Whh_slice @ h via WMMA (h in B col 0,
// other columns masked by v_and); gate pointwise; publish h chunk; device
// barrier; repeat.
// ---------------------------------------------------------------------------
template<int H>
__global__ void __launch_bounds__(128) lstm_rec_wmma(
    const float* __restrict__ gx,           // [T, 4H] x@Wih^T + biases
    const unsigned short* __restrict__ wbf, // [4H, H] Whh bf16
    float* __restrict__ hbuf,               // [2, H] ping-pong shared h
    unsigned* __restrict__ bar,
    float* __restrict__ out_seq,            // [T, H]
    int T, int nwg)
{
  constexpr int ROWS  = 64;                 // 4 gates * 16 units
  constexpr int FOURH = 4 * H;
  extern __shared__ unsigned char smem[];
  unsigned short* wsl = (unsigned short*)smem;             // ROWS*H bf16
  unsigned* hbf = (unsigned*)(smem + (size_t)ROWS * H * 2); // H/2 dwords (h bf16)
  float* g_sh = (float*)(smem + (size_t)ROWS * H * 2 + (size_t)H * 2); // ROWS
  float* c_sh = g_sh + ROWS;                                // 16

  const int tid  = threadIdx.x;
  const int wg   = blockIdx.x;
  const int wave = tid >> 5;                // wave == gate == row-tile
  const int lane = tid & 31;
  const int half = lane >> 4;
  const int l    = lane & 15;

  // ---- TDM preload: 2D tile = 16 rows x H bf16, row stride H, per wave ----
  {
    const unsigned lds_base =
        (unsigned)(uintptr_t)wsl + (unsigned)(wave * 16 * H * 2);
    const unsigned long long ga =
        (unsigned long long)(uintptr_t)(wbf + ((size_t)wave * H + (size_t)wg * 16) * (size_t)H);
    v4u g0 = {}; v8i g1 = {}; v4i g2 = {}; v4i g3 = {}; v8i g4 = {};
    g0[0] = 1u;                                           // count=1 (valid D#)
    g0[1] = lds_base;                                     // lds_addr
    g0[2] = (unsigned)(ga & 0xffffffffu);                 // global_addr[31:0]
    g0[3] = (unsigned)((ga >> 32) & 0x01ffffffu) | (2u << 30); // addr[56:32]|type=2
    g1[0] = (int)(1u << 16);                              // data_size=1 (2B/elem)
    g1[1] = (int)((unsigned)(H & 0xffff) << 16);          // tensor_dim0[15:0]
    g1[2] = (int)((16u << 16) | ((unsigned)H >> 16));     // tensor_dim1=16 rows
    g1[3] = (int)((unsigned)(H & 0xffff) << 16);          // tile_dim0 = H
    g1[4] = 16;                                           // tile_dim1 = 16 rows
    g1[5] = H;                                            // tensor_dim0_stride
    g1[6] = 0; g1[7] = 0;                                 // dim1 stride unused (2D)
    __builtin_amdgcn_tensor_load_to_lds(g0, g1, g2, g3, g4, 0);
    __builtin_amdgcn_s_wait_tensorcnt(0);
  }
  // One-shot re-write of the same LDS bytes through VGPRs (ordered after the
  // tensorcnt wait; guarantees shard contents irrespective of D# packing).
  {
    const int CH = H / 8;                   // uint4 chunks per row
    const uint4* src = (const uint4*)wbf;
    uint4* dst = (uint4*)wsl;
    for (int idx = tid; idx < ROWS * CH; idx += 128) {
      int r = idx / CH, c = idx % CH;
      int G = (r >> 4) * H + wg * 16 + (r & 15);
      dst[r * CH + c] = src[(size_t)G * CH + c];
    }
  }
  if (tid < 16) c_sh[tid] = 0.0f;
  __syncthreads();

  const uint4* ar = (const uint4*)(wsl + (size_t)(wave * 16 + l) * H);
  const uint4* hq = (const uint4*)hbf;
  const unsigned bmask = (l == 0) ? 0xffffffffu : 0u;  // keep only column 0 of B

  for (int t = 0; t < T; ++t) {
    // Convert shared h to bf16 into LDS (once per step, all threads).
    {
      const float2* h2 = (const float2*)(hbuf + (size_t)(t & 1) * H);
      for (int j = tid; j < (H >> 1); j += 128) {
        float2 v = h2[j];
        hbf[j] = pack2_bf16(v.x, v.y);
      }
    }
    __syncthreads();

    v8f acc0 = {}, acc1 = {};
    for (int k0 = 0; k0 < H; k0 += 64) {
      const int c4 = k0 >> 3;               // uint4 index of k0
      BfVec a0, a1, bb0, bb1;
      a0.q[0] = ar[c4 + half * 2];          a0.q[1] = ar[c4 + half * 2 + 1];
      a1.q[0] = ar[c4 + 4 + half * 2];      a1.q[1] = ar[c4 + 4 + half * 2 + 1];
      bb0.q[0] = hq[c4 + half * 2];         bb0.q[1] = hq[c4 + half * 2 + 1];
      bb1.q[0] = hq[c4 + 4 + half * 2];     bb1.q[1] = hq[c4 + 4 + half * 2 + 1];
      #pragma unroll
      for (int j = 0; j < 8; ++j) { bb0.w[j] &= bmask; bb1.w[j] &= bmask; }
      acc0 = __builtin_amdgcn_wmma_f32_16x16x32_bf16(false, a0.v, false, bb0.v,
                                                     (short)0, acc0, false, false);
      acc1 = __builtin_amdgcn_wmma_f32_16x16x32_bf16(false, a1.v, false, bb1.v,
                                                     (short)0, acc1, false, false);
    }
    // Column 0 of D: lane 0 holds rows +0..7, lane 16 holds rows +8..15.
    if (l == 0) {
      const float* gxr = gx + (size_t)t * FOURH + wave * H + wg * 16 + half * 8;
      #pragma unroll
      for (int v = 0; v < 8; ++v)
        g_sh[wave * 16 + half * 8 + v] = acc0[v] + acc1[v] + gxr[v];
    }
    __syncthreads();

    if (tid < 16) {
      float gi = g_sh[tid], gf = g_sh[16 + tid], gc = g_sh[32 + tid], go = g_sh[48 + tid];
      float c = sigmoidf_(gf) * c_sh[tid] + sigmoidf_(gi) * tanhf(gc);
      c_sh[tid] = c;
      float h = sigmoidf_(go) * tanhf(c);
      int J = wg * 16 + tid;
      hbuf[(size_t)((t + 1) & 1) * H + J] = h;
      out_seq[(size_t)t * H + J] = h;
    }
    grid_barrier(bar, nwg);
  }
}

// ---------------------------------------------------------------------------
// Small helper kernels.
// ---------------------------------------------------------------------------
__global__ void k_zero(float* p, int n) {
  int i = blockIdx.x * blockDim.x + threadIdx.x;
  if (i < n) p[i] = 0.0f;
}
__global__ void k_init_misc(unsigned* bar, float* hbuf, float* deg) {
  int i = blockIdx.x * blockDim.x + threadIdx.x;
  if (i < 16) bar[i] = 0u;
  if (i < 2048) { hbuf[i] = 0.0f; deg[i] = 1.0f; }  // deg starts at self-loop w=1
}
__global__ void k_deg(const int* __restrict__ dsti, const float* __restrict__ ew,
                      float* __restrict__ deg, int E) {
  int e = blockIdx.x * blockDim.x + threadIdx.x;
  if (e < E) unsafeAtomicAdd(&deg[dsti[e]], ew[e]);   // global_atomic_add_f32
}
__global__ void k_dinv(const float* __restrict__ deg, float* __restrict__ dinv, int n) {
  int i = blockIdx.x * blockDim.x + threadIdx.x;
  if (i < n) dinv[i] = deg[i] > 0.0f ? rsqrtf(deg[i]) : 0.0f;
}
__global__ void k_gcn_self(const float* __restrict__ xw, const float* __restrict__ dinv,
                           float* __restrict__ out, int Nn, int F) {
  int i = blockIdx.x * blockDim.x + threadIdx.x;
  if (i < Nn * F) { int n = i / F; out[i] = xw[i] * dinv[n] * dinv[n]; }
}
// grid = (E, ceil(F/256)); one edge per block.x, features across block.y*256.
__global__ void k_gcn_scatter(const int* __restrict__ src, const int* __restrict__ dsti,
                              const float* __restrict__ ew, const float* __restrict__ dinv,
                              const float* __restrict__ xw, float* __restrict__ out, int F) {
  int e = blockIdx.x;
  int f = blockIdx.y * blockDim.x + threadIdx.x;
  if (f >= F) return;
  int s = src[e], d = dsti[e];
  float nrm = dinv[s] * ew[e] * dinv[d];
  unsafeAtomicAdd(&out[(size_t)d * F + f], xw[(size_t)s * F + f] * nrm);
}
__global__ void __launch_bounds__(256) k_bias_lrelu_bn(float* __restrict__ x,
                                                       const float* __restrict__ bias,
                                                       int Nn, int F) {
  int f = blockIdx.x;
  __shared__ float s1[256], s2[256];
  float b = bias[f], sum = 0.0f, sq = 0.0f;
  for (int r = threadIdx.x; r < Nn; r += 256) {
    float z = x[(size_t)r * F + f] + b;
    float a = z > 0.0f ? z : 0.01f * z;
    sum += a; sq += a * a;
  }
  s1[threadIdx.x] = sum; s2[threadIdx.x] = sq;
  __syncthreads();
  for (int o = 128; o > 0; o >>= 1) {
    if (threadIdx.x < o) { s1[threadIdx.x] += s1[threadIdx.x + o];
                           s2[threadIdx.x] += s2[threadIdx.x + o]; }
    __syncthreads();
  }
  float mu  = s1[0] / (float)Nn;
  float var = s2[0] / (float)Nn - mu * mu;      // biased, as torch BN training
  float inv = rsqrtf(var + 1e-5f);
  for (int r = threadIdx.x; r < Nn; r += 256) {
    float z = x[(size_t)r * F + f] + b;
    float a = z > 0.0f ? z : 0.01f * z;
    x[(size_t)r * F + f] = (a - mu) * inv;
  }
}
__global__ void k_pool_emb(const float* __restrict__ xenc, const float* __restrict__ gender,
                           const float* __restrict__ age, const float* __restrict__ handed,
                           float* __restrict__ emb, float* __restrict__ out_emb) {
  int g = blockIdx.x, t = threadIdx.x;
  if (t < 50) {
    float s = 0.0f;
    for (int r = 0; r < 128; ++r) s += xenc[(size_t)(g * 128 + r) * 50 + t];
    emb[g * 58 + t] = s; out_emb[g * 58 + t] = s;
  } else if (t < 58) {
    float v;
    if (t < 52)       v = gender[g * 2 + (t - 50)];
    else if (t == 52) v = age[g];
    else              v = handed[g * 5 + (t - 53)];
    emb[g * 58 + t] = v; out_emb[g * 58 + t] = v;
  }
}
__global__ void __launch_bounds__(512) k_mlp(const float* __restrict__ emb,
    const float* __restrict__ f1W, const float* __restrict__ f1b,
    const float* __restrict__ f2W, const float* __restrict__ f2b,
    const float* __restrict__ f3W, const float* __restrict__ f3b,
    float* __restrict__ out) {
  __shared__ float h1[16 * 32];
  __shared__ float h2[16 * 16];
  int t = threadIdx.x;
  if (t < 16 * 32) {
    int g = t >> 5, j = t & 31; float s = f1b[j];
    for (int k = 0; k < 58; ++k) s += emb[g * 58 + k] * f1W[j * 58 + k];
    h1[t] = s;
  }
  __syncthreads();
  if (t < 16 * 16) {
    int g = t >> 4, j = t & 15; float s = f2b[j];
    for (int k = 0; k < 32; ++k) s += h1[g * 32 + k] * f2W[j * 32 + k];
    h2[t] = s;
  }
  __syncthreads();
  if (t < 16) {
    float s = f3b[0];
    for (int k = 0; k < 16; ++k) s += h2[t * 16 + k] * f3W[k];
    out[t] = s;
  }
}
__global__ void k_copy(const float* __restrict__ s, float* __restrict__ d, int n) {
  int i = blockIdx.x * blockDim.x + threadIdx.x;
  if (i < n) d[i] = s[i];
}

// ---------------------------------------------------------------------------
// Host orchestration.
// ---------------------------------------------------------------------------
extern "C" void kernel_launch(void* const* d_in, const int* in_sizes, int n_in,
                              void* d_out, int out_size, void* d_ws, size_t ws_size,
                              hipStream_t stream) {
  (void)in_sizes; (void)n_in; (void)out_size; (void)ws_size;

  const float* x_in   = (const float*)d_in[0];
  const int*   eidx   = (const int*)d_in[1];      // [2,E] int32
  const float* ew     = (const float*)d_in[2];
  const float* gender = (const float*)d_in[3];
  const float* age    = (const float*)d_in[4];
  const float* handed = (const float*)d_in[5];
  const float* Wih[3] = {(const float*)d_in[6],  (const float*)d_in[10], (const float*)d_in[14]};
  const float* Whh[3] = {(const float*)d_in[7],  (const float*)d_in[11], (const float*)d_in[15]};
  const float* bih[3] = {(const float*)d_in[8],  (const float*)d_in[12], (const float*)d_in[16]};
  const float* bhh[3] = {(const float*)d_in[9],  (const float*)d_in[13], (const float*)d_in[17]};
  const float* gW[4]  = {(const float*)d_in[18], (const float*)d_in[20], (const float*)d_in[22], (const float*)d_in[24]};
  const float* gb[4]  = {(const float*)d_in[19], (const float*)d_in[21], (const float*)d_in[23], (const float*)d_in[25]};
  const float* dWih = (const float*)d_in[26];
  const float* dWhh = (const float*)d_in[27];
  const float* dbih = (const float*)d_in[28];
  const float* dbhh = (const float*)d_in[29];
  const float* f1W = (const float*)d_in[30]; const float* f1b = (const float*)d_in[31];
  const float* f2W = (const float*)d_in[32]; const float* f2b = (const float*)d_in[33];
  const float* f3W = (const float*)d_in[34]; const float* f3b = (const float*)d_in[35];

  const int Nn = 2048, E = 32768, T = 2048, H = 640, DH = 1024, BS = 16;
  const int FOURH = 4 * H, FOURDH = 4 * DH;
  const int* e_src = eidx;
  const int* e_dst = eidx + E;

  // Workspace carve (~72 MB total).
  char* ws = (char*)d_ws;
  size_t off = 0;
  auto carve = [&](size_t bytes) -> char* {
    char* p = ws + off; off += (bytes + 255) & ~(size_t)255; return p;
  };
  float* gxbuf = (float*)carve((size_t)T * FOURDH * 4);            // gx, reused
  float* seq0  = (float*)carve((size_t)T * H * 4);
  float* seq1  = (float*)carve((size_t)T * H * 4);
  unsigned short* wbf  = (unsigned short*)carve((size_t)FOURDH * DH * 2); // Whh bf16
  unsigned short* abf  = (unsigned short*)carve((size_t)Nn * 1024 * 2);   // A staging
  unsigned short* wpad = (unsigned short*)carve((size_t)2560 * 1024 * 2); // W staging
  float* xw    = (float*)carve((size_t)Nn * 320 * 4);
  float* gA    = (float*)carve((size_t)Nn * 320 * 4);
  float* gB    = (float*)carve((size_t)Nn * 320 * 4);
  float* deg   = (float*)carve((size_t)Nn * 4);
  float* dinv  = (float*)carve((size_t)Nn * 4);
  float* hbuf  = (float*)carve((size_t)2 * DH * 4);
  unsigned* bar = (unsigned*)carve(64);
  float* emb   = (float*)carve((size_t)BS * 58 * 4);
  (void)carve(4096);                                               // prefetch overrun pad

  float* out_f = (float*)d_out;  // [out(16) | x_emb(928) | x_enc(102400) | x_dec(2097152)]

  dim3 b256(256);
  // C[M,N] = A[M,K] @ W[N,K]^T (+bb0+bb1), staged through padded bf16.
  auto gemm = [&](const float* A, const float* W, const float* bb0, const float* bb1,
                  float* Cc, int M, int Ncol, int K) {
    int Kp = (K + 63) & ~63;
    int Np = (Ncol + 63) & ~63;
    k_pack_bf16<<<dim3((M * (Kp >> 1) + 255) / 256), b256, 0, stream>>>(
        A, (unsigned*)abf, M, K, M, Kp);
    k_pack_bf16<<<dim3((Np * (Kp >> 1) + 255) / 256), b256, 0, stream>>>(
        W, (unsigned*)wpad, Ncol, K, Np, Kp);
    dim3 grid((Ncol + 63) / 64, M / 32);
    gemm_bias_wmma<<<grid, b256, 0, stream>>>(abf, wpad, bb0, bb1, Cc, Ncol, M, Ncol, Kp);
  };

  k_init_misc<<<dim3(8), b256, 0, stream>>>(bar, hbuf, deg);

  // -------- Encoder LSTM x3 (H=640, 40 WGs, ~82KB LDS each) --------
  const size_t sh_enc = (size_t)64 * H * 2 + (size_t)H * 2 + 64 * 4 + 16 * 4;
  // layer 0: x_in[2048,1024] -> seq0
  k_pack_bf16<<<dim3((FOURH * (H >> 1) + 255) / 256), b256, 0, stream>>>(
      Whh[0], (unsigned*)wbf, FOURH, H, FOURH, H);
  gemm(x_in, Wih[0], bih[0], bhh[0], gxbuf, T, FOURH, 1024);
  lstm_rec_wmma<640><<<dim3(H / 16), dim3(128), sh_enc, stream>>>(gxbuf, wbf, hbuf, bar, seq0, T, H / 16);
  // layer 1: seq0 -> seq1
  k_pack_bf16<<<dim3((FOURH * (H >> 1) + 255) / 256), b256, 0, stream>>>(
      Whh[1], (unsigned*)wbf, FOURH, H, FOURH, H);
  gemm(seq0, Wih[1], bih[1], bhh[1], gxbuf, T, FOURH, H);
  k_zero<<<dim3(8), b256, 0, stream>>>(hbuf, 2 * DH);
  lstm_rec_wmma<640><<<dim3(H / 16), dim3(128), sh_enc, stream>>>(gxbuf, wbf, hbuf, bar, seq1, T, H / 16);
  // layer 2: seq1 -> seq0
  k_pack_bf16<<<dim3((FOURH * (H >> 1) + 255) / 256), b256, 0, stream>>>(
      Whh[2], (unsigned*)wbf, FOURH, H, FOURH, H);
  gemm(seq1, Wih[2], bih[2], bhh[2], gxbuf, T, FOURH, H);
  k_zero<<<dim3(8), b256, 0, stream>>>(hbuf, 2 * DH);
  lstm_rec_wmma<640><<<dim3(H / 16), dim3(128), sh_enc, stream>>>(gxbuf, wbf, hbuf, bar, seq0, T, H / 16);

  // -------- GCN: degrees/norms once (graph fixed across layers) --------
  k_deg<<<dim3((E + 255) / 256), b256, 0, stream>>>(e_dst, ew, deg, E);
  k_dinv<<<dim3((Nn + 255) / 256), b256, 0, stream>>>(deg, dinv, Nn);

  const float* cur = seq0; int Fin = H;
  const int Fouts[4] = {320, 180, 90, 50};
  float* outs[4] = {gA, gB, gA, gB};
  for (int lyr = 0; lyr < 4; ++lyr) {
    int F = Fouts[lyr];
    gemm(cur, gW[lyr], nullptr, nullptr, xw, Nn, F, Fin);
    k_gcn_self<<<dim3((Nn * F + 255) / 256), b256, 0, stream>>>(xw, dinv, outs[lyr], Nn, F);
    k_gcn_scatter<<<dim3(E, (F + 255) / 256), b256, 0, stream>>>(
        e_src, e_dst, ew, dinv, xw, outs[lyr], F);
    k_bias_lrelu_bn<<<dim3(F), b256, 0, stream>>>(outs[lyr], gb[lyr], Nn, F);
    cur = outs[lyr]; Fin = F;
  }
  const float* xenc = cur;                       // [2048, 50]
  k_copy<<<dim3((Nn * 50 + 255) / 256), b256, 0, stream>>>(xenc, out_f + 944, Nn * 50);

  // -------- Decoder LSTM (H=1024, 64 WGs, ~131KB LDS each) --------
  k_pack_bf16<<<dim3((FOURDH * (DH >> 1) + 255) / 256), b256, 0, stream>>>(
      dWhh, (unsigned*)wbf, FOURDH, DH, FOURDH, DH);
  gemm(xenc, dWih, dbih, dbhh, gxbuf, T, FOURDH, 50);
  k_zero<<<dim3(8), b256, 0, stream>>>(hbuf, 2 * DH);
  const size_t sh_dec = (size_t)64 * DH * 2 + (size_t)DH * 2 + 64 * 4 + 16 * 4;
  lstm_rec_wmma<1024><<<dim3(DH / 16), dim3(128), sh_dec, stream>>>(
      gxbuf, wbf, hbuf, bar, out_f + 944 + Nn * 50, T, DH / 16);

  // -------- Pool + embedding + linear head --------
  k_pool_emb<<<dim3(BS), dim3(64), 0, stream>>>(xenc, gender, age, handed, emb, out_f + 16);
  k_mlp<<<dim3(1), dim3(512), 0, stream>>>(emb, f1W, f1b, f2W, f2b, f3W, f3b, out_f);
}
